// RecurrentRGCN_78658031058993
// MI455X (gfx1250) — compile-verified, matrix operand order checked by reading, not run on hardware
//
#include <hip/hip_runtime.h>
#include <math.h>

typedef __attribute__((ext_vector_type(16))) _Float16 v16h;
typedef __attribute__((ext_vector_type(8)))  float    v8f;

#define N_ENT 20000
#define N_REL 460
#define N_RELP 480   // padded K for attn GEMM
#define H     128
#define NE    400000
#define BB    1024
#define TT    8
#define RRELU_SLOPE_C 0.2291666666666667f

// ---------------- WMMA fragment helpers (gfx1250, wave32) ----------------
// 16-bit A-matrix 16x32 packing (ISA 7.12.2): lane-half lh holds
// K = lh*8 + {0..7} in elements 0..7 and K = 16 + lh*8 + {0..7} in elements 8..15.
// Both runs are contiguous -> 4x global_load_b128 per fragment.
__device__ __forceinline__ v16h load_frag_rows(const float* __restrict__ P, int ld,
                                               int base_m, int base_k, int maxr, int lane) {
  int m = base_m + (lane & 15);
  if (m > maxr - 1) m = maxr - 1;
  int lh = lane >> 4;
  const float* p = P + m * ld + base_k + lh * 8;
  float4 q0 = *(const float4*)(p);
  float4 q1 = *(const float4*)(p + 4);
  float4 q2 = *(const float4*)(p + 16);
  float4 q3 = *(const float4*)(p + 20);
  v16h f;
  f[0]  = (_Float16)q0.x; f[1]  = (_Float16)q0.y; f[2]  = (_Float16)q0.z; f[3]  = (_Float16)q0.w;
  f[4]  = (_Float16)q1.x; f[5]  = (_Float16)q1.y; f[6]  = (_Float16)q1.z; f[7]  = (_Float16)q1.w;
  f[8]  = (_Float16)q2.x; f[9]  = (_Float16)q2.y; f[10] = (_Float16)q2.z; f[11] = (_Float16)q2.w;
  f[12] = (_Float16)q3.x; f[13] = (_Float16)q3.y; f[14] = (_Float16)q3.z; f[15] = (_Float16)q3.w;
  return f;
}

__device__ __forceinline__ v8f wmma_f16(v16h a, v16h b, v8f c) {
  return __builtin_amdgcn_wmma_f32_16x16x32_f16(false, a, false, b, (short)0, c, false, false);
}

// ---------------- Kernels ----------------

// out[n*ldo + m] = in[m*ldi + n] for m < Mr, else 0   (out is Nn x Mpad)
__global__ void transpose_pad_kernel(const float* __restrict__ in, float* __restrict__ out,
                                     int Mr, int Nn, int ldi, int Mpad) {
  int idx = blockIdx.x * blockDim.x + threadIdx.x;
  if (idx >= Nn * Mpad) return;
  int n = idx / Mpad, m = idx - n * Mpad;
  out[idx] = (m < Mr) ? in[m * ldi + n] : 0.f;
}

// Scatter (h[src]+rel[type]) into pre_agg[dst], count degrees.
__global__ void edge_scatter_kernel(const float* __restrict__ h, const float* __restrict__ rel,
                                    const int* __restrict__ src, const int* __restrict__ dst,
                                    const int* __restrict__ typ,
                                    float* __restrict__ pre_agg, float* __restrict__ deg, int E) {
  int e = blockIdx.x;
  if (e >= E) return;
  int s = src[e], d = dst[e], t = typ[e];
  int j = threadIdx.x;
  atomicAdd(&pre_agg[d * H + j], h[s * H + j] + rel[t * H + j]);
  if (j == 0) atomicAdd(&deg[d], 1.0f);
}

// Fused: agg = pre_agg@Wn ; loop = h@{Wl|We} per deg ; pre_emb = rrelu(agg*norm + loop)
// Weight matrices passed pre-transposed (N x K row-major), so all loads are row-contiguous.
__global__ void node_update_kernel(const float* __restrict__ pre_agg, const float* __restrict__ h,
                                   const float* __restrict__ deg,
                                   const float* __restrict__ WnT, const float* __restrict__ WlT,
                                   const float* __restrict__ WeT, float* __restrict__ pre_emb) {
  int lane = threadIdx.x;
  int base_m = blockIdx.x * 16;
  int lh = lane >> 4, col15 = lane & 15;
  for (int nt = 0; nt < 8; ++nt) {
    v8f accA = {}, accL = {}, accE = {};
    for (int kt = 0; kt < 4; ++kt) {
      v16h aA = load_frag_rows(pre_agg, H, base_m, kt * 32, N_ENT, lane);
      v16h aH = load_frag_rows(h,       H, base_m, kt * 32, N_ENT, lane);
      v16h bN = load_frag_rows(WnT, H, nt * 16, kt * 32, H, lane);
      accA = wmma_f16(aA, bN, accA);
      v16h bL = load_frag_rows(WlT, H, nt * 16, kt * 32, H, lane);
      accL = wmma_f16(aH, bL, accL);
      v16h bE = load_frag_rows(WeT, H, nt * 16, kt * 32, H, lane);
      accE = wmma_f16(aH, bE, accE);
    }
    int col = nt * 16 + col15;
#pragma unroll
    for (int r = 0; r < 8; ++r) {
      int row = base_m + (lh ? 8 : 0) + r;
      float d = deg[row];
      float norm = d > 0.f ? 1.f / d : 0.f;
      float x = accA[r] * norm + (d > 0.f ? accL[r] : accE[r]);
      pre_emb[row * H + col] = x >= 0.f ? x : x * RRELU_SLOPE_C;
    }
  }
}

// out = alpha*(X @ W^T + bias) + resid    (X: MxK row-major, W: NxK row-major)
// Each wave computes a 64x16 tile: the weight (B) fragment is loaded once per K-step
// and reused across 4 row-subtiles (4 accumulators) -> 4x less weight traffic.
__global__ void lin_wmma_kernel(const float* __restrict__ X, const float* __restrict__ W,
                                const float* __restrict__ bias, const float* __restrict__ resid,
                                float* __restrict__ out, int M, int N, int K, int ldo, float alpha) {
  int lane = threadIdx.x;
  int base_m = blockIdx.x * 64, base_n = blockIdx.y * 16;
  v8f acc[4] = {{}, {}, {}, {}};
  for (int kt = 0; kt < K; kt += 32) {
    v16h b = load_frag_rows(W, K, base_n, kt, N, lane);
#pragma unroll
    for (int i = 0; i < 4; ++i) {
      v16h a = load_frag_rows(X, K, base_m + i * 16, kt, M, lane);
      acc[i] = wmma_f16(a, b, acc[i]);
    }
  }
  int col = base_n + (lane & 15);
  float bv = (bias && col < N) ? bias[col] : 0.f;
#pragma unroll
  for (int i = 0; i < 4; ++i) {
    int rbase = base_m + i * 16 + ((lane >> 4) ? 8 : 0);
#pragma unroll
    for (int r = 0; r < 8; ++r) {
      int row = rbase + r;
      if (row < M && col < N) {
        float v = (acc[i][r] + bv) * alpha;
        if (resid) v += resid[row * ldo + col];
        out[row * ldo + col] = v;
      }
    }
  }
}

// attn[b, 0:480] = padded masked-softmax( hist[t][b,:] * A[b,:] )
__global__ void softmax_attn_kernel(const float* __restrict__ hist_t, const float* __restrict__ A,
                                    float* __restrict__ attn) {
  int b = blockIdx.x, tid = threadIdx.x;
  __shared__ float red[256];
  int r0 = tid, r1 = tid + 256;
  bool ok0 = r0 < N_REL, ok1 = r1 < N_REL;
  float x0 = 0.f, x1 = 0.f, v0 = -1e30f, v1 = -1e30f;
  if (ok0) { x0 = hist_t[b * (TT * N_REL) + r0] * A[b * N_RELP + r0];
             if (x0 == 0.f) x0 = -1e9f; v0 = x0; }
  if (ok1) { x1 = hist_t[b * (TT * N_REL) + r1] * A[b * N_RELP + r1];
             if (x1 == 0.f) x1 = -1e9f; v1 = x1; }
  red[tid] = fmaxf(v0, v1); __syncthreads();
  for (int s = 128; s > 0; s >>= 1) { if (tid < s) red[tid] = fmaxf(red[tid], red[tid + s]); __syncthreads(); }
  float m = red[0]; __syncthreads();
  float e0 = ok0 ? expf(x0 - m) : 0.f;
  float e1 = ok1 ? expf(x1 - m) : 0.f;
  red[tid] = e0 + e1; __syncthreads();
  for (int s = 128; s > 0; s >>= 1) { if (tid < s) red[tid] += red[tid + s]; __syncthreads(); }
  float inv = 1.f / red[0];
  attn[b * N_RELP + r0] = ok0 ? e0 * inv : 0.f;
  if (r1 < N_RELP) attn[b * N_RELP + r1] = ok1 ? e1 * inv : 0.f;
}

__global__ void gru_gate_kernel(const float* __restrict__ gi, const float* __restrict__ gh,
                                const float* __restrict__ hin, float* __restrict__ hout) {
  int idx = blockIdx.x * blockDim.x + threadIdx.x;
  if (idx >= BB * H) return;
  int b = idx >> 7, j = idx & 127;
  const float* gib = gi + b * (3 * H);
  const float* ghb = gh + b * (3 * H);
  float r = 1.f / (1.f + expf(-(gib[j] + ghb[j])));
  float z = 1.f / (1.f + expf(-(gib[H + j] + ghb[H + j])));
  float n = tanhf(gib[2 * H + j] + r * ghb[2 * H + j]);
  hout[idx] = (1.f - z) * n + z * hin[idx];
}

__global__ void gather_rows_kernel(const float* __restrict__ src, const int* __restrict__ tri,
                                   int comp, float* __restrict__ out) {
  int b = blockIdx.x;
  int idx = tri[b * 3 + comp];
  out[b * H + threadIdx.x] = src[idx * H + threadIdx.x];
}

__global__ void mul_kernel(const float* __restrict__ a, const float* __restrict__ b,
                           float* __restrict__ o, int n) {
  int i = blockIdx.x * blockDim.x + threadIdx.x;
  if (i < n) o[i] = a[i] * b[i];
}

__global__ void mse_kernel(const float* __restrict__ a, const float* __restrict__ b,
                           float* __restrict__ acc, int n) {
  __shared__ float red[256];
  int i = blockIdx.x * 256 + threadIdx.x;
  float v = 0.f;
  if (i < n) { float d = a[i] - b[i]; v = d * d; }
  red[threadIdx.x] = v; __syncthreads();
  for (int s = 128; s > 0; s >>= 1) { if (threadIdx.x < s) red[threadIdx.x] += red[threadIdx.x + s]; __syncthreads(); }
  if (threadIdx.x == 0) atomicAdd(acc, red[0]);
}

// score_ = lhs @ aligned_all^T ; score = sigmoid(score_) ; accumulate sum(softplus(x)).
// Each wave computes a 16x64 tile: A-fragment reused across 4 n-tiles.
__global__ void score_wmma_kernel(const float* __restrict__ lhs, const float* __restrict__ AA,
                                  float* __restrict__ score, float* __restrict__ nll_acc) {
  int lane = threadIdx.x;
  int base_m = blockIdx.x * 16;
  int base_n = blockIdx.y * 64;
  v8f acc[4] = {{}, {}, {}, {}};
#pragma unroll
  for (int kt = 0; kt < H; kt += 32) {
    v16h a = load_frag_rows(lhs, H, base_m, kt, BB, lane);
#pragma unroll
    for (int j = 0; j < 4; ++j) {
      v16h b = load_frag_rows(AA, H, base_n + j * 16, kt, N_ENT, lane);
      acc[j] = wmma_f16(a, b, acc[j]);
    }
  }
  int rbase = base_m + ((lane >> 4) ? 8 : 0);
  float nll = 0.f;
#pragma unroll
  for (int j = 0; j < 4; ++j) {
    int col = base_n + j * 16 + (lane & 15);
    if (col < N_ENT) {
#pragma unroll
      for (int r = 0; r < 8; ++r) {
        float x = acc[j][r];
        score[(rbase + r) * N_ENT + col] = 1.f / (1.f + expf(-x));
        nll += (x > 0.f) ? (x + log1pf(expf(-x))) : log1pf(expf(x));  // -log_sigmoid(-x)
      }
    }
  }
  for (int off = 16; off > 0; off >>= 1) nll += __shfl_down(nll, off);
  if (lane == 0) atomicAdd(nll_acc, nll);
}

// per-row correction: log1mp[obj]-logp[obj] == -score_[b,obj]
__global__ void score_corr_kernel(const float* __restrict__ lhs, const float* __restrict__ AA,
                                  const int* __restrict__ tri, float* __restrict__ corr_acc) {
  int b = blockIdx.x;
  int obj = tri[b * 3 + 2];
  __shared__ float red[128];
  red[threadIdx.x] = lhs[b * H + threadIdx.x] * AA[obj * H + threadIdx.x];
  __syncthreads();
  for (int s = 64; s > 0; s >>= 1) { if (threadIdx.x < s) red[threadIdx.x] += red[threadIdx.x + s]; __syncthreads(); }
  if (threadIdx.x == 0) atomicAdd(corr_acc, -red[0]);
}

__global__ void finalize_kernel(const float* __restrict__ acc, float* __restrict__ out_losses) {
  out_losses[0] = acc[0] / (float)(BB * H);
  out_losses[1] = (acc[1] + acc[2]) / (float)((size_t)BB * N_ENT);
}

// ---------------- Launch ----------------
extern "C" void kernel_launch(void* const* d_in, const int* in_sizes, int n_in,
                              void* d_out, int out_size, void* d_ws, size_t ws_size,
                              hipStream_t stream) {
  const float* ent_emb   = (const float*)d_in[0];
  const float* rel_emb   = (const float*)d_in[1];
  const float* Wn        = (const float*)d_in[2];
  const float* Wl        = (const float*)d_in[3];
  const float* We        = (const float*)d_in[4];
  const float* w_ih      = (const float*)d_in[5];
  const float* w_hh      = (const float*)d_in[6];
  const float* b_ih      = (const float*)d_in[7];
  const float* b_hh      = (const float*)d_in[8];
  const float* W_map1    = (const float*)d_in[9];
  const float* b_map1    = (const float*)d_in[10];
  const float* W_map2    = (const float*)d_in[11];
  const float* b_map2    = (const float*)d_in[12];
  const float* W_h1      = (const float*)d_in[13];
  const float* b_h1      = (const float*)d_in[14];
  const float* W_h2      = (const float*)d_in[15];
  const float* b_h2      = (const float*)d_in[16];
  const float* W_align   = (const float*)d_in[17];
  const float* b_align   = (const float*)d_in[18];
  const float* W_attn    = (const float*)d_in[19];
  const float* b_attn    = (const float*)d_in[20];
  const float* partial   = (const float*)d_in[21];
  const int*   edge_src  = (const int*)d_in[22];
  const int*   edge_dst  = (const int*)d_in[23];
  const int*   edge_type = (const int*)d_in[24];
  const int*   triples   = (const int*)d_in[25];

  float* score  = (float*)d_out;
  float* losses = score + (size_t)BB * N_ENT;

  // workspace carve-out
  float* ws = (float*)d_ws;
  size_t o = 0;
  auto alloc = [&](size_t n) { float* p = ws + o; o += n; return p; };
  float* pre_agg     = alloc((size_t)N_ENT * H);
  float* deg         = alloc(N_ENT);
  float* pre_emb     = alloc((size_t)N_ENT * H);
  float* rel_mid     = alloc((size_t)N_REL * 256);
  float* mapped_rel  = alloc((size_t)N_REL * H);
  float* mappedT     = alloc((size_t)H * N_RELP);   // 128 x 480, zero padded
  float* WnT         = alloc((size_t)H * H);
  float* WlT         = alloc((size_t)H * H);
  float* WeT         = alloc((size_t)H * H);
  float* q_rel       = alloc((size_t)BB * H);
  float* attn_q      = alloc((size_t)BB * H);
  float* Abuf        = alloc((size_t)BB * N_RELP);
  float* attn        = alloc((size_t)BB * N_RELP);
  float* rel_path    = alloc((size_t)BB * H);
  float* gru_h       = alloc((size_t)BB * H);
  float* gi          = alloc((size_t)BB * 3 * H);
  float* gh          = alloc((size_t)BB * 3 * H);
  float* h1          = alloc((size_t)BB * H);
  float* pred        = alloc((size_t)BB * H);
  float* gh2         = alloc((size_t)BB * H);
  float* sub_emb     = alloc((size_t)BB * H);
  float* aligned_sub = alloc((size_t)BB * H);
  float* lhs         = alloc((size_t)BB * H);
  float* acc         = alloc(4);
  float* aligned_all = pre_agg;  // reuse: pre_agg dead after node_update

  hipMemsetAsync(pre_agg, 0, (size_t)N_ENT * H * sizeof(float), stream);
  hipMemsetAsync(deg,     0, N_ENT * sizeof(float), stream);
  hipMemsetAsync(gru_h,   0, (size_t)BB * H * sizeof(float), stream);
  hipMemsetAsync(acc,     0, 4 * sizeof(float), stream);

  // 0) pre-transpose weights so every WMMA operand load is row-contiguous
  transpose_pad_kernel<<<(H * H + 255) / 256, 256, 0, stream>>>(Wn, WnT, H, H, H, H);
  transpose_pad_kernel<<<(H * H + 255) / 256, 256, 0, stream>>>(Wl, WlT, H, H, H, H);
  transpose_pad_kernel<<<(H * H + 255) / 256, 256, 0, stream>>>(We, WeT, H, H, H, H);

  // 1) message passing: scatter then one GEMM (linearity of @W_n)
  edge_scatter_kernel<<<NE, H, 0, stream>>>(ent_emb, rel_emb, edge_src, edge_dst, edge_type,
                                            pre_agg, deg, NE);
  node_update_kernel<<<N_ENT / 16, 32, 0, stream>>>(pre_agg, ent_emb, deg, WnT, WlT, WeT, pre_emb);

  // 2) mapped_rel = lin(lin(rel_emb, W_map1), W_map2); transpose+pad for the scan GEMM
  lin_wmma_kernel<<<dim3(8, 16), 32, 0, stream>>>(rel_emb, W_map1, b_map1, nullptr,
                                                  rel_mid, N_REL, 256, H, 256, 1.f);
  lin_wmma_kernel<<<dim3(8, 8), 32, 0, stream>>>(rel_mid, W_map2, b_map2, nullptr,
                                                 mapped_rel, N_REL, H, 256, H, 1.f);
  transpose_pad_kernel<<<(H * N_RELP + 255) / 256, 256, 0, stream>>>(mapped_rel, mappedT,
                                                                     N_REL, H, H, N_RELP);
  gather_rows_kernel<<<BB, H, 0, stream>>>(mapped_rel, triples, 1, q_rel);

  // 3) A = lin(q_rel, W_attn) @ mapped_rel^T   (stored with ld=480)
  lin_wmma_kernel<<<dim3(BB / 64, 8), 32, 0, stream>>>(q_rel, W_attn, b_attn, nullptr,
                                                       attn_q, BB, H, H, H, 1.f);
  lin_wmma_kernel<<<dim3(BB / 64, 29), 32, 0, stream>>>(attn_q, mapped_rel, nullptr, nullptr,
                                                        Abuf, BB, N_REL, H, N_RELP, 1.f);

  // 4) GRU scan over T steps
  for (int t = 0; t < TT; ++t) {
    softmax_attn_kernel<<<BB, 256, 0, stream>>>(partial + (size_t)t * N_REL, Abuf, attn);
    // rel_path = attn @ mapped_rel == lin(attn, mappedT) with K=480 (attn tail zeroed)
    lin_wmma_kernel<<<dim3(BB / 64, 8), 32, 0, stream>>>(attn, mappedT, nullptr, nullptr,
                                                         rel_path, BB, H, N_RELP, H, 1.f);
    lin_wmma_kernel<<<dim3(BB / 64, 24), 32, 0, stream>>>(rel_path, w_ih, b_ih, nullptr,
                                                          gi, BB, 3 * H, H, 3 * H, 1.f);
    lin_wmma_kernel<<<dim3(BB / 64, 24), 32, 0, stream>>>(gru_h, w_hh, b_hh, nullptr,
                                                          gh, BB, 3 * H, H, 3 * H, 1.f);
    gru_gate_kernel<<<(BB * H) / 256, 256, 0, stream>>>(gi, gh, gru_h, gru_h);
  }

  // 5) predicted_hist = 0.1*lin(lin(q_rel,W_h1),W_h2) + q_rel ; match_loss
  lin_wmma_kernel<<<dim3(BB / 64, 8), 32, 0, stream>>>(q_rel, W_h1, b_h1, nullptr,
                                                       h1, BB, H, H, H, 1.f);
  lin_wmma_kernel<<<dim3(BB / 64, 8), 32, 0, stream>>>(h1, W_h2, b_h2, q_rel,
                                                       pred, BB, H, H, H, 0.1f);
  mse_kernel<<<(BB * H) / 256, 256, 0, stream>>>(pred, gru_h, acc + 0, BB * H);

  // 6) gh2 = gru_cell(q_rel, predicted_hist)
  lin_wmma_kernel<<<dim3(BB / 64, 24), 32, 0, stream>>>(q_rel, w_ih, b_ih, nullptr,
                                                        gi, BB, 3 * H, H, 3 * H, 1.f);
  lin_wmma_kernel<<<dim3(BB / 64, 24), 32, 0, stream>>>(pred, w_hh, b_hh, nullptr,
                                                        gh, BB, 3 * H, H, 3 * H, 1.f);
  gru_gate_kernel<<<(BB * H) / 256, 256, 0, stream>>>(gi, gh, pred, gh2);

  // 7) scoring
  gather_rows_kernel<<<BB, H, 0, stream>>>(pre_emb, triples, 0, sub_emb);
  lin_wmma_kernel<<<dim3((N_ENT + 63) / 64, 8), 32, 0, stream>>>(pre_emb, W_align, b_align, nullptr,
                                                                 aligned_all, N_ENT, H, H, H, 1.f);
  lin_wmma_kernel<<<dim3(BB / 64, 8), 32, 0, stream>>>(sub_emb, W_align, b_align, nullptr,
                                                       aligned_sub, BB, H, H, H, 1.f);
  mul_kernel<<<(BB * H) / 256, 256, 0, stream>>>(aligned_sub, gh2, lhs, BB * H);
  score_wmma_kernel<<<dim3(BB / 16, (N_ENT + 63) / 64), 32, 0, stream>>>(lhs, aligned_all,
                                                                         score, acc + 1);
  score_corr_kernel<<<BB, H, 0, stream>>>(lhs, aligned_all, triples, acc + 2);
  finalize_kernel<<<1, 1, 0, stream>>>(acc, losses);
}
